// Head_2680059593146
// MI455X (gfx1250) — compile-verified, matrix operand order checked by reading, not run on hardware
//
#include <hip/hip_runtime.h>
#include <hip/hip_bf16.h>

// Causal single-head attention, B=8 T=2048 C=768 HS=64.
// k0: W -> W^T f16      k1: x @ W -> q,k f16 [BT,64], v^T f16 [B,64,T]
// k2: flash attention: TDM (tensor_load_to_lds) double-buffered K/V tiles,
//     v_wmma_f32_16x16x32_f16, online softmax, P transposed through LDS.

typedef __attribute__((ext_vector_type(16))) _Float16 v16h;
typedef __attribute__((ext_vector_type(8)))  _Float16 v8h;
typedef __attribute__((ext_vector_type(8)))  float    v8f;
typedef __attribute__((ext_vector_type(4)))  unsigned int v4u32;
typedef __attribute__((ext_vector_type(8)))  int          v8i32;
typedef __attribute__((ext_vector_type(4)))  int          v4i32;

#define BT    16384      // B*T
#define TT    2048
#define CC    768
#define HS    64
#define SCALE 0.036084391824351615f   // C**-0.5

#if __has_builtin(__builtin_amdgcn_tensor_load_to_lds) && \
    __has_builtin(__builtin_amdgcn_s_wait_tensorcnt)
#define HAVE_TDM 1
#else
#define HAVE_TDM 0
#endif

// Lane's 16 fragment halves = two contiguous 16B runs at +8g and +16+8g halves
// (16-bit A 16x32 layout, ISA 7.12.2: K(j) = j + 8g + (j>=8 ? 8 : 0)).
__device__ __forceinline__ v16h load_frag16(const _Float16* base, int g) {
    v8h lo = *(const v8h*)(base + 8 * g);
    v8h hi = *(const v8h*)(base + 16 + 8 * g);
    return __builtin_shufflevector(lo, hi, 0, 1, 2, 3, 4, 5, 6, 7,
                                           8, 9, 10, 11, 12, 13, 14, 15);
}

#if HAVE_TDM
// Issue one 2-D f16 tile DMA: global (row-major, stride0 elems) -> LDS packed.
// D# bitfields per CDNA5 ISA ch.8 (group0 128b, group1 256b, groups2/3 zero).
// Toolchain here = clang-23 / therock-10.0: 6-arg builtin form.
__device__ __forceinline__ void tdm_load_tile(
    const _Float16* gbase, unsigned int lds_off,
    unsigned int tdim0, unsigned int tdim1,
    unsigned int tile0, unsigned int tile1, unsigned int stride0)
{
    unsigned long long ga = (unsigned long long)(size_t)gbase;
    v4u32 g0;
    g0[0] = 1u;                                            // count=1, user D#
    g0[1] = lds_off;                                       // lds_addr [63:32]
    g0[2] = (unsigned int)ga;                              // global_addr lo
    g0[3] = ((unsigned int)(ga >> 32) & 0x1FFFFFFu)        // global_addr[56:32]
            | (2u << 30);                                  // type=2 (image)
    v8i32 g1;
    g1[0] = (int)(1u << 16);                               // data_size=1 (2B)
    g1[1] = (int)((tdim0 & 0xFFFFu) << 16);                // tensor_dim0[15:0]
    g1[2] = (int)((tdim0 >> 16) | ((tdim1 & 0xFFFFu) << 16));
    g1[3] = (int)((tdim1 >> 16) | (tile0 << 16));          // tile_dim0
    g1[4] = (int)(tile1 & 0xFFFFu);                        // tile_dim1 (dim2=0)
    g1[5] = (int)stride0;                                  // dim0_stride lo32
    g1[6] = 0;                                             // stride hi, dim1_stride
    g1[7] = 0;
    v4i32 z4 = {0, 0, 0, 0};
    v8i32 z8 = {0, 0, 0, 0, 0, 0, 0, 0};
    __builtin_amdgcn_tensor_load_to_lds(g0, g1, z4, z4, z8, 0);
}
#endif

// ---------------------------------------------------------------------------
// Kernel 0: W [C,HS] f32 -> W^T [HS,C] f16 for q,k,v.  (tiny: 0.3 MB)
// ---------------------------------------------------------------------------
__global__ __launch_bounds__(256) void wtrans(
    const float* __restrict__ Wq, const float* __restrict__ Wk,
    const float* __restrict__ Wv, _Float16* __restrict__ wt)
{
    const int idx   = blockIdx.x * 256 + threadIdx.x;
    const int which = idx / (CC * HS);
    const int rem   = idx % (CC * HS);
    const int c = rem >> 6, h = rem & 63;
    const float* __restrict__ W = (which == 0) ? Wq : (which == 1) ? Wk : Wv;
    wt[(size_t)which * CC * HS + (size_t)h * CC + c] = (_Float16)W[rem];
}

// ---------------------------------------------------------------------------
// Kernel 1: QKV projection. grid = (BT/16, 3), block = 128 (4 waves).
// ---------------------------------------------------------------------------
__global__ __launch_bounds__(128) void qkv_proj(
    const float*    __restrict__ x,
    const _Float16* __restrict__ wt,   // [3, HS, C] f16
    _Float16* __restrict__ qo,         // [BT, 64]
    _Float16* __restrict__ ko,         // [BT, 64]
    _Float16* __restrict__ vto)        // [B, 64, T]
{
    const int lane  = threadIdx.x & 31;
    const int wave  = threadIdx.x >> 5;
    const int m     = lane & 15;
    const int g     = lane >> 4;
    const int row0  = blockIdx.x * 16;
    const int which = blockIdx.y;
    const int col   = wave * 16 + m;

    const float*    __restrict__ xrow = x  + (size_t)(row0 + m) * CC;
    const _Float16* __restrict__ wrow = wt + (size_t)which * CC * HS
                                           + (size_t)col * CC;
    v8f acc = {};
    for (int cc = 0; cc < CC; cc += 32) {
        v8f alo = *(const v8f*)(xrow + cc + 8 * g);
        v8f ahi = *(const v8f*)(xrow + cc + 16 + 8 * g);
        v8h alo_h = __builtin_convertvector(alo, v8h);
        v8h ahi_h = __builtin_convertvector(ahi, v8h);
        v16h a = __builtin_shufflevector(alo_h, ahi_h, 0, 1, 2, 3, 4, 5, 6, 7,
                                                       8, 9, 10, 11, 12, 13, 14, 15);
        v16h b = load_frag16(wrow + cc, g);
        acc = __builtin_amdgcn_wmma_f32_16x16x32_f16(
                  false, a, false, b, (short)0, acc, false, false);
    }

    if (which == 2) {
        const int b_idx = row0 >> 11;
#pragma unroll
        for (int r = 0; r < 8; ++r) {
            const int grow = row0 + r + 8 * g;
            const int t    = grow & (TT - 1);
            vto[((size_t)b_idx * HS + col) * TT + t] = (_Float16)acc[r];
        }
    } else {
        _Float16* __restrict__ dst = (which == 0) ? qo : ko;
#pragma unroll
        for (int r = 0; r < 8; ++r)
            dst[(size_t)(row0 + r + 8 * g) * HS + col] = (_Float16)acc[r];
    }
}

// ---------------------------------------------------------------------------
// Kernel 2: flash attention. grid = BT/64 = 256 blocks, block = 128 (4 waves).
// Each wave owns one 16-query-row tile; 32 keys per iteration.
// K tile [32,64] and V tile [64,32] are TDM-DMA'd to LDS, double buffered.
// ---------------------------------------------------------------------------
__global__ __launch_bounds__(128) void attn(
    const _Float16* __restrict__ q,     // [BT, 64]
    const _Float16* __restrict__ k,     // [BT, 64]
    const _Float16* __restrict__ vt,    // [B, 64, T]
    float* __restrict__ out)            // [BT, 64]
{
#if HAVE_TDM
    __shared__ __align__(16) _Float16 lds_k[4][2][32 * 64];  // 32 KB
    __shared__ __align__(16) _Float16 lds_v[4][2][64 * 32];  // 32 KB
#endif
    __shared__ __align__(16) _Float16 lds_p[4][16 * 32];     // 4 KB

    const int lane = threadIdx.x & 31;
    const int wave = threadIdx.x >> 5;
    const int m    = lane & 15;
    const int g    = lane >> 4;

    const int blk   = blockIdx.x;
    const int b_idx = blk >> 5;                    // 32 blocks per batch
    const int qtile = (blk & 31) * 4 + wave;       // 0..127
    const int q0    = qtile * 16;

    const _Float16* __restrict__ qb = q  + (size_t)b_idx * TT * HS;
    const _Float16* __restrict__ kb = k  + (size_t)b_idx * TT * HS;
    const _Float16* __restrict__ vb = vt + (size_t)b_idx * HS * TT;
    _Float16* __restrict__ ldsp = &lds_p[wave][0];

    // Q fragments: rows q0+m, two K-dim chunks of 32 (2x b128 each)
    v16h qa[2];
#pragma unroll
    for (int c = 0; c < 2; ++c)
        qa[c] = load_frag16(qb + (size_t)(q0 + m) * HS + c * 32, g);

    v8f o0 = {}, o1 = {}, o2 = {}, o3 = {};
    float mrow[8], lrow[8];
#pragma unroll
    for (int r = 0; r < 8; ++r) { mrow[r] = -__builtin_inff(); lrow[r] = 0.0f; }

    const int kend = q0 + 16;                      // causal key count

#if HAVE_TDM
    const unsigned int koff[2] = {
        (unsigned int)(size_t)&lds_k[wave][0][0],
        (unsigned int)(size_t)&lds_k[wave][1][0] };
    const unsigned int voff[2] = {
        (unsigned int)(size_t)&lds_v[wave][0][0],
        (unsigned int)(size_t)&lds_v[wave][1][0] };
    // preload tile 0: K rows [0,32) x 64 dims; V dims [0,64) x keys [0,32)
    tdm_load_tile(kb, koff[0], HS, TT, HS, 32, HS);
    tdm_load_tile(vb, voff[0], TT, HS, 32, HS, TT);
#endif

    for (int k0 = 0; k0 < kend; k0 += 32) {
#if HAVE_TDM
        const int buf = (k0 >> 5) & 1;
        const _Float16* __restrict__ ldsk = &lds_k[wave][buf][0];
        const _Float16* __restrict__ ldsv = &lds_v[wave][buf][0];
        if (k0 + 32 < kend) {                       // issue next pair, keep 2 in flight
            tdm_load_tile(kb + (size_t)(k0 + 32) * HS, koff[buf ^ 1], HS, TT, HS, 32, HS);
            tdm_load_tile(vb + (k0 + 32),              voff[buf ^ 1], TT, HS, 32, HS, TT);
            __builtin_amdgcn_s_wait_tensorcnt(2);
        } else {
            __builtin_amdgcn_s_wait_tensorcnt(0);
        }
#else
        __builtin_prefetch(kb + (size_t)(k0 + 32 + m) * HS, 0, 1);
        __builtin_prefetch(vb + (size_t)(m) * TT + k0 + 32, 0, 1);
#endif

        // ---- S = Q @ K^T for two 16-key tiles ----
        v8f s0 = {}, s1 = {};
#pragma unroll
        for (int c = 0; c < 2; ++c) {
#if HAVE_TDM
            v16h bf0 = load_frag16(ldsk + (size_t)(m)      * HS + c * 32, g);
            v16h bf1 = load_frag16(ldsk + (size_t)(16 + m) * HS + c * 32, g);
#else
            v16h bf0 = load_frag16(kb + (size_t)(k0 + m)      * HS + c * 32, g);
            v16h bf1 = load_frag16(kb + (size_t)(k0 + 16 + m) * HS + c * 32, g);
#endif
            s0 = __builtin_amdgcn_wmma_f32_16x16x32_f16(
                     false, qa[c], false, bf0, (short)0, s0, false, false);
            s1 = __builtin_amdgcn_wmma_f32_16x16x32_f16(
                     false, qa[c], false, bf1, (short)0, s1, false, false);
        }

        // ---- online softmax (S row = r + 8g, key col = t*16 + m) ----
#pragma unroll
        for (int r = 0; r < 8; ++r) {
            const int qrow = q0 + r + 8 * g;
            float v0 = (k0 + m      <= qrow) ? s0[r] * SCALE : -__builtin_inff();
            float v1 = (k0 + 16 + m <= qrow) ? s1[r] * SCALE : -__builtin_inff();

            float mx = fmaxf(v0, v1);
#pragma unroll
            for (int d = 1; d < 16; d <<= 1) mx = fmaxf(mx, __shfl_xor(mx, d, 32));

            const float mnew = fmaxf(mrow[r], mx);
            const float p0   = __expf(v0 - mnew);
            const float p1   = __expf(v1 - mnew);
            const float corr = __expf(mrow[r] - mnew);

            float rs = p0 + p1;
#pragma unroll
            for (int d = 1; d < 16; d <<= 1) rs += __shfl_xor(rs, d, 32);

            lrow[r] = lrow[r] * corr + rs;
            mrow[r] = mnew;
            o0[r] *= corr; o1[r] *= corr; o2[r] *= corr; o3[r] *= corr;

            const int srow = r + 8 * g;
            ldsp[srow * 32 + m]      = (_Float16)p0;
            ldsp[srow * 32 + 16 + m] = (_Float16)p1;
        }

        asm volatile("s_wait_dscnt 0" ::: "memory");   // LDS writes visible wave-wide

        // ---- P as A-fragment (row = m, K = key): 2x ds_load_b128 ----
        v16h pa = load_frag16(ldsp + m * 32, g);

        // ---- O += P @ V : 4 N-tiles over HS ----
#if HAVE_TDM
        v16h vf0 = load_frag16(ldsv + (size_t)(0 * 16 + m) * 32, g);
        v16h vf1 = load_frag16(ldsv + (size_t)(1 * 16 + m) * 32, g);
        v16h vf2 = load_frag16(ldsv + (size_t)(2 * 16 + m) * 32, g);
        v16h vf3 = load_frag16(ldsv + (size_t)(3 * 16 + m) * 32, g);
#else
        v16h vf0 = load_frag16(vb + (size_t)(0 * 16 + m) * TT + k0, g);
        v16h vf1 = load_frag16(vb + (size_t)(1 * 16 + m) * TT + k0, g);
        v16h vf2 = load_frag16(vb + (size_t)(2 * 16 + m) * TT + k0, g);
        v16h vf3 = load_frag16(vb + (size_t)(3 * 16 + m) * TT + k0, g);
#endif
        o0 = __builtin_amdgcn_wmma_f32_16x16x32_f16(
                 false, pa, false, vf0, (short)0, o0, false, false);
        o1 = __builtin_amdgcn_wmma_f32_16x16x32_f16(
                 false, pa, false, vf1, (short)0, o1, false, false);
        o2 = __builtin_amdgcn_wmma_f32_16x16x32_f16(
                 false, pa, false, vf2, (short)0, o2, false, false);
        o3 = __builtin_amdgcn_wmma_f32_16x16x32_f16(
                 false, pa, false, vf3, (short)0, o3, false, false);
    }

    // ---- normalize & store ----
    float* __restrict__ ob = out + ((size_t)b_idx * TT + q0) * HS;
#pragma unroll
    for (int r = 0; r < 8; ++r) {
        const float inv = 1.0f / lrow[r];
        const int row   = r + 8 * g;
        ob[(size_t)row * HS + 0 * 16 + m] = o0[r] * inv;
        ob[(size_t)row * HS + 1 * 16 + m] = o1[r] * inv;
        ob[(size_t)row * HS + 2 * 16 + m] = o2[r] * inv;
        ob[(size_t)row * HS + 3 * 16 + m] = o3[r] * inv;
    }
}

// ---------------------------------------------------------------------------
extern "C" void kernel_launch(void* const* d_in, const int* in_sizes, int n_in,
                              void* d_out, int out_size, void* d_ws, size_t ws_size,
                              hipStream_t stream) {
    const float* x  = (const float*)d_in[0];
    const float* Wk = (const float*)d_in[1];
    const float* Wq = (const float*)d_in[2];
    const float* Wv = (const float*)d_in[3];
    float* out = (float*)d_out;

    _Float16* qw  = (_Float16*)d_ws;                 // [BT,64]   2 MB
    _Float16* kw  = qw  + (size_t)BT * HS;           // [BT,64]   2 MB
    _Float16* vtw = kw  + (size_t)BT * HS;           // [B,64,T]  2 MB
    _Float16* wtw = vtw + (size_t)BT * HS;           // [3,HS,C]  0.3 MB

    wtrans<<<dim3(3 * CC * HS / 256), 256, 0, stream>>>(Wq, Wk, Wv, wtw);

    dim3 g1(BT / 16, 3);
    qkv_proj<<<g1, 128, 0, stream>>>(x, wtw, qw, kw, vtw);

    dim3 g2(BT / 64);
    attn<<<g2, 128, 0, stream>>>(qw, kw, vtw, out);
}
